// MultiHeadSelfDotAttention_27178553049200
// MI455X (gfx1250) — matrix operand order repaired and checked by reading, not compiled, not run on hardware
//
#include <hip/hip_runtime.h>

// ---------------------------------------------------------------------------
// MultiHeadSelfDotAttention (linear attention) for gfx1250 / MI455X.
// All heavy math via V_WMMA_F32_16X16X32_F16; tiles staged to LDS with
// GLOBAL_LOAD_ASYNC_TO_LDS_B128 when available. All operands layout-native:
// A is [M][K], B is [N][K] (pre-transposed once), so staging is pure B128
// copies and every fragment read is 1-2 contiguous 16B LDS loads.
// ---------------------------------------------------------------------------

#define HEADS 8
#define NSEQ  8192
#define DIN   512
#define DIMF  512

typedef __attribute__((ext_vector_type(16))) _Float16 v16h;
typedef __attribute__((ext_vector_type(8)))  _Float16 v8h;
typedef __attribute__((ext_vector_type(4)))  _Float16 v4h;
typedef __attribute__((ext_vector_type(8)))  float    v8f;
typedef __attribute__((ext_vector_type(4)))  int      v4i;

static constexpr int LDSS = 40;  // padded LDS row stride (f16); 80B rows keep 16B align

#if __has_builtin(__builtin_amdgcn_global_load_async_to_lds_b128)
#define USE_ASYNC_LDS 1
typedef __attribute__((address_space(1))) v4i* global_v4i;
typedef __attribute__((address_space(3))) v4i* lds_v4i;
#endif

// stage 32 bytes (16 f16) global -> LDS
__device__ __forceinline__ void stage32B(const _Float16* __restrict__ gp,
                                         _Float16* __restrict__ sp) {
#ifdef USE_ASYNC_LDS
    __builtin_amdgcn_global_load_async_to_lds_b128(
        (global_v4i)gp, (lds_v4i)sp, 0, 0);
    __builtin_amdgcn_global_load_async_to_lds_b128(
        (global_v4i)(gp + 8), (lds_v4i)(sp + 8), 0, 0);
#else
    *(v8h*)sp       = *(const v8h*)gp;
    *(v8h*)(sp + 8) = *(const v8h*)(gp + 8);
#endif
}

__device__ __forceinline__ void wait_stage() {
#ifdef USE_ASYNC_LDS
#if __has_builtin(__builtin_amdgcn_s_wait_asynccnt)
    __builtin_amdgcn_s_wait_asynccnt(0);
#else
    asm volatile("s_wait_asynccnt 0x0" ::: "memory");
#endif
#endif
    __syncthreads();
}

// ---------------------------------------------------------------------------
// fp32 -> f16 conversion, 4 elements per thread (for x)
// ---------------------------------------------------------------------------
__global__ __launch_bounds__(256)
void cvt_f32_f16(const float* __restrict__ in, _Float16* __restrict__ out) {
    size_t i = (size_t)blockIdx.x * 256 + threadIdx.x;
    float4 v = ((const float4*)in)[i];
    v4h o;
    o[0] = (_Float16)v.x; o[1] = (_Float16)v.y;
    o[2] = (_Float16)v.z; o[3] = (_Float16)v.w;
    ((v4h*)out)[i] = o;
}

// fp32 [H][512][512] (d,e) -> f16 transposed [H][512][512] (e,d); one-time
__global__ __launch_bounds__(256)
void cvt_f32_f16_T(const float* __restrict__ in, _Float16* __restrict__ out) {
    const int h = blockIdx.z;
    const int d = blockIdx.y;
    const int e = blockIdx.x * 256 + threadIdx.x;
    out[((size_t)h * DIMF + e) * DIN + d] =
        (_Float16)in[((size_t)h * DIN + d) * DIMF + e];
}

// ---------------------------------------------------------------------------
// Tiled WMMA GEMM: C[b] = alpha * A[b] @ B[b]^T-stored  (B given as [N][K])
//   optional per-M-row divide by rowSum, optional transposed (col-major) C.
// Block: 128 threads = 4 waves; block tile 128x128, BK = 32 (== WMMA K).
// Each wave: 4x4 grid of 16x16 WMMA tiles (16 v_wmma per K-step).
// ---------------------------------------------------------------------------
template<bool OUT_F16, bool ROW_SCALE, bool OUT_TRANS>
__global__ __launch_bounds__(128)
void gemm_wmma(const _Float16* __restrict__ Ag, const _Float16* __restrict__ Bg,
               void* __restrict__ Cg, const float* __restrict__ rowSum,
               int M, int N, int K,
               long long aBatch, long long bBatch, long long cBatch, int rsBatch,
               float alpha)
{
    __shared__ _Float16 ldsA[128 * LDSS];
    __shared__ _Float16 ldsB[128 * LDSS];

    const int b = blockIdx.z;
    const _Float16* A = Ag + (size_t)b * aBatch;   // [M][K] row-major
    const _Float16* B = Bg + (size_t)b * bBatch;   // [N][K] row-major

    const int nBase = blockIdx.x * 128;
    const int mBase = blockIdx.y * 128;

    const int tid   = threadIdx.x;
    const int lane  = tid & 31;
    const int wid   = tid >> 5;
    const int waveM = wid >> 1;        // 0..1
    const int waveN = wid & 1;         // 0..1
    const int lg    = lane >> 4;       // lane group 0/1
    const int l16   = lane & 15;

    const int srow = tid >> 1;         // 0..63 (staging row)
    const int sseg = tid & 1;          // 0..1  (staging 16-f16 segment)

    v8f zero = {0.f,0.f,0.f,0.f,0.f,0.f,0.f,0.f};
    v8f acc[4][4];
    #pragma unroll
    for (int tm = 0; tm < 4; ++tm)
        #pragma unroll
        for (int tn = 0; tn < 4; ++tn) acc[tm][tn] = zero;

    for (int k0 = 0; k0 < K; k0 += 32) {
        // ---- stage A and B tiles (128 x 32 f16 each), pure B128 copies ----
        #pragma unroll
        for (int it = 0; it < 2; ++it) {
            int row = srow + it * 64;
            stage32B(A + (size_t)(mBase + row) * K + k0 + sseg * 16,
                     ldsA + row * LDSS + sseg * 16);
            stage32B(B + (size_t)(nBase + row) * K + k0 + sseg * 16,
                     ldsB + row * LDSS + sseg * 16);
        }
        wait_stage();

        // prefetch next K-step (global_prefetch_b8)
        if (k0 + 32 < K) {
            __builtin_prefetch(A + (size_t)(mBase + srow) * K + k0 + 32, 0, 0);
            __builtin_prefetch(B + (size_t)(nBase + srow) * K + k0 + 32, 0, 0);
        }

        // ---- fragments: documented 16-bit A layout; B lane=col, K contig ----
        v16h aF[4], bF[4];
        #pragma unroll
        for (int t = 0; t < 4; ++t) {
            {
                int m = waveM * 64 + t * 16 + l16;
                const _Float16* p = ldsA + m * LDSS + lg * 8;  // K = lg*8+{0..7}, 16+lg*8+{0..7}
                v8h lo = *(const v8h*)p;
                v8h hi = *(const v8h*)(p + 16);
                #pragma unroll
                for (int i = 0; i < 8; ++i) { aF[t][i] = lo[i]; aF[t][i + 8] = hi[i]; }
            }
            {
                int n = waveN * 64 + t * 16 + l16;
                const _Float16* p = ldsB + n * LDSS + lg * 8;  // K = lg*8+{0..7}, 16+lg*8+{0..7}
                v8h lo = *(const v8h*)p;
                v8h hi = *(const v8h*)(p + 16);
                #pragma unroll
                for (int i = 0; i < 8; ++i) { bF[t][i] = lo[i]; bF[t][i + 8] = hi[i]; }
            }
        }

        // ---- 16 WMMAs per K-step ----
        #pragma unroll
        for (int tm = 0; tm < 4; ++tm)
            #pragma unroll
            for (int tn = 0; tn < 4; ++tn)
                acc[tm][tn] = __builtin_amdgcn_wmma_f32_16x16x32_f16(
                    false, aF[tm], false, bF[tn],
                    (short)0, acc[tm][tn], false, false);

        __syncthreads();
    }

    // ---- epilogue (C layout: VGPR j -> M = j + 8*lg, N = lane&15) ----
    _Float16* C16 = (_Float16*)Cg;
    float*    C32 = (float*)Cg;
    #pragma unroll
    for (int tm = 0; tm < 4; ++tm) {
        int rowBase = mBase + waveM * 64 + tm * 16 + 8 * lg;
        #pragma unroll
        for (int tn = 0; tn < 4; ++tn) {
            int col = nBase + waveN * 64 + tn * 16 + l16;
            if constexpr (OUT_TRANS) {
                // 8 consecutive M-rows live in one lane -> one contiguous v8h store
                v8h pack;
                #pragma unroll
                for (int j = 0; j < 8; ++j) {
                    float v = acc[tm][tn][j] * alpha;
                    if constexpr (ROW_SCALE)
                        v /= rowSum[(size_t)b * rsBatch + rowBase + j];
                    pack[j] = (_Float16)v;
                }
                *(v8h*)(C16 + (size_t)b * cBatch + (size_t)col * M + rowBase) = pack;
            } else {
                #pragma unroll
                for (int j = 0; j < 8; ++j) {
                    int row = rowBase + j;
                    float v = acc[tm][tn][j] * alpha;
                    if constexpr (ROW_SCALE)
                        v /= rowSum[(size_t)b * rsBatch + row];
                    size_t idx = (size_t)b * cBatch + (size_t)row * N + col;
                    if constexpr (OUT_F16) C16[idx] = (_Float16)v;
                    else                   C32[idx] = v;
                }
            }
        }
    }
}

// ---------------------------------------------------------------------------
// Column softmax over the seq axis for Q (row-major [n][e]).
// Writes unnormalized exp() in place and column sums S_q[h][e].
// ---------------------------------------------------------------------------
__global__ __launch_bounds__(256)
void softmax_col(_Float16* __restrict__ Q, float* __restrict__ Sq) {
    const int h   = blockIdx.y;
    const int col = blockIdx.x * 256 + threadIdx.x;
    _Float16* q = Q + (size_t)h * NSEQ * DIMF;

    float mx = -3.0e38f;
    for (int n = 0; n < NSEQ; ++n)
        mx = fmaxf(mx, (float)q[(size_t)n * DIMF + col]);

    float s = 0.f;
    for (int n = 0; n < NSEQ; ++n) {
        size_t idx = (size_t)n * DIMF + col;
        float p = __expf((float)q[idx] - mx);
        s += p;
        q[idx] = (_Float16)p;
    }
    Sq[h * DIMF + col] = s;
}

// ---------------------------------------------------------------------------
// K softmax (over feature axis) on transposed Kt [e][n]: per column n,
// reduce over the 512 e-rows (fully coalesced across threads), write exp in
// place, and fold the denominator into Vt [f][n] columns.
// ---------------------------------------------------------------------------
__global__ __launch_bounds__(256)
void softmax_colT_scale_v(_Float16* __restrict__ Kt, _Float16* __restrict__ Vt) {
    const int h = blockIdx.y;
    const size_t n = (size_t)blockIdx.x * 256 + threadIdx.x;
    _Float16* kp = Kt + (size_t)h * DIMF * NSEQ;
    _Float16* vp = Vt + (size_t)h * DIMF * NSEQ;

    float mx = -3.0e38f;
    for (int e = 0; e < DIMF; ++e)
        mx = fmaxf(mx, (float)kp[(size_t)e * NSEQ + n]);

    float s = 0.f;
    for (int e = 0; e < DIMF; ++e) {
        size_t idx = (size_t)e * NSEQ + n;
        float p = __expf((float)kp[idx] - mx);
        s += p;
        kp[idx] = (_Float16)p;
    }
    const float inv = 1.0f / s;
    for (int f = 0; f < DIMF; ++f) {
        size_t idx = (size_t)f * NSEQ + n;
        vp[idx] = (_Float16)((float)vp[idx] * inv);
    }
}

// ---------------------------------------------------------------------------
// Host-side orchestration
// ---------------------------------------------------------------------------
extern "C" void kernel_launch(void* const* d_in, const int* in_sizes, int n_in,
                              void* d_out, int out_size, void* d_ws, size_t ws_size,
                              hipStream_t stream) {
    (void)in_sizes; (void)n_in; (void)out_size; (void)ws_size;

    const float* x  = (const float*)d_in[0];
    const float* WQ = (const float*)d_in[1];
    const float* WK = (const float*)d_in[2];
    const float* WV = (const float*)d_in[3];
    const float* WO = (const float*)d_in[4];
    float* out = (float*)d_out;

    char* ws = (char*)d_ws;
    const size_t SZ_XH  = (size_t)NSEQ * DIN * 2;            // 8 MiB
    const size_t SZ_W   = (size_t)HEADS * DIN * DIMF * 2;    // 4 MiB each
    const size_t SZ_ACT = (size_t)HEADS * NSEQ * DIMF * 2;   // 64 MiB each
    const size_t SZ_SQ  = (size_t)HEADS * DIMF * 4;          // 16 KiB
    const size_t SZ_CTX = (size_t)HEADS * DIMF * DIMF * 2;   // 4 MiB

    size_t off = 0;
    _Float16* xh    = (_Float16*)(ws + off); off += SZ_XH;
    _Float16* wqt   = (_Float16*)(ws + off); off += SZ_W;    // WQ^T [e][d]
    _Float16* wkt   = (_Float16*)(ws + off); off += SZ_W;
    _Float16* wvt   = (_Float16*)(ws + off); off += SZ_W;
    _Float16* wot   = (_Float16*)(ws + off); off += SZ_W;    // WO^T [f][g]
    _Float16* Qs    = (_Float16*)(ws + off); off += SZ_ACT;  // [n][e] -> exp(Q)
    _Float16* Kt    = (_Float16*)(ws + off); off += SZ_ACT;  // [e][n] -> exp(K)
    _Float16* Vt    = (_Float16*)(ws + off); off += SZ_ACT;  // [f][n] -> V/S_k
    float*    Sq    = (float*)   (ws + off); off += SZ_SQ;
    _Float16* ctx   = (_Float16*)(ws + off); off += SZ_CTX;  // [e][f] = rk^T V
    _Float16* ctx3t = (_Float16*)(ws + off); off += SZ_CTX;  // [f][e] = (ctx@WO)/S_q

    const float scale = 0.04419417382415922f;  // 1/sqrt(512)
    const long long HW = (long long)DIN * DIMF;       // per-head 512x512 stride
    const long long HN = (long long)NSEQ * DIMF;      // per-head 8192x512 stride

    // 1) fp32 -> f16 conversions; weights transposed to [N][K] once
    cvt_f32_f16<<<(NSEQ * DIN) / 1024, 256, 0, stream>>>(x, xh);
    dim3 gT(DIMF / 256, DIN, HEADS);
    cvt_f32_f16_T<<<gT, 256, 0, stream>>>(WQ, wqt);
    cvt_f32_f16_T<<<gT, 256, 0, stream>>>(WK, wkt);
    cvt_f32_f16_T<<<gT, 256, 0, stream>>>(WV, wvt);
    cvt_f32_f16_T<<<gT, 256, 0, stream>>>(WO, wot);

    // 2) projections (128x128 block tiles)
    dim3 gProj(DIMF / 128, NSEQ / 128, HEADS);
    // Qs[n][e] = scale * x@WQ  (plain epilogue)
    gemm_wmma<true, false, false><<<gProj, 128, 0, stream>>>(
        xh, wqt, Qs, nullptr, NSEQ, DIMF, DIN, 0, HW, HN, 0, scale);
    // Kt[e][n] = scale * x@WK  (transposed epilogue, contiguous v8h stores)
    gemm_wmma<true, false, true><<<gProj, 128, 0, stream>>>(
        xh, wkt, Kt, nullptr, NSEQ, DIMF, DIN, 0, HW, HN, 0, scale);
    // Vt[f][n] = x@WV          (transposed epilogue)
    gemm_wmma<true, false, true><<<gProj, 128, 0, stream>>>(
        xh, wvt, Vt, nullptr, NSEQ, DIMF, DIN, 0, HW, HN, 0, 1.0f);

    // 3) softmaxes: exp in place, denominators folded forward
    softmax_col<<<dim3(DIMF / 256, HEADS), 256, 0, stream>>>(Qs, Sq);
    softmax_colT_scale_v<<<dim3(NSEQ / 256, HEADS), 256, 0, stream>>>(Kt, Vt);

    // 4) ctx[e][f] = exp(K)^T @ (V/S_k): A=Kt [512][8192], B=Vt [512][8192]
    dim3 gCtx(DIMF / 128, DIMF / 128, HEADS);
    gemm_wmma<true, false, false><<<gCtx, 128, 0, stream>>>(
        Kt, Vt, ctx, nullptr, DIMF, DIMF, NSEQ, HN, HN, HW, 0, 1.0f);

    // 5) ctx3t[f][e] = (ctx @ WO)[e][f] / S_q[e]: A=ctx, B=wot, trans+rowscale
    gemm_wmma<true, true, true><<<gCtx, 128, 0, stream>>>(
        ctx, wot, ctx3t, Sq, DIMF, DIMF, DIMF, HW, HW, HW, DIMF, 1.0f);

    // 6) out[n][f] = exp(Q) @ ctx3t^T-stored: A=Qs, B=ctx3t, f32 out
    gemm_wmma<false, false, false><<<gProj, 128, 0, stream>>>(
        Qs, ctx3t, out, nullptr, NSEQ, DIMF, DIN, HN, HW, HN, 0, 1.0f);
}
